// MultiHeadAttention_45913200394352
// MI455X (gfx1250) — matrix operand order repaired and checked by reading, not compiled, not run on hardware
//
#include <hip/hip_runtime.h>
#include <hip/hip_bf16.h>

typedef __bf16 bf16_t;
typedef __attribute__((ext_vector_type(16))) __bf16 v16bf;
typedef __attribute__((ext_vector_type(8)))  __bf16 v8bf;
typedef __attribute__((ext_vector_type(8)))  float  v8f;

#define DMODEL 1024
#define NHEADS 16
#define DK     64
#define SLEN   2048
#define BATCH  4
#define MTOT   (BATCH * SLEN)   // 8192 tokens

// ---------------------------------------------------------------------------
// Fragment loaders following CDNA5 ISA 7.12.2 layouts (wave32).
// A-matrix 16x32 bf16: lanes 0-15 -> M=lane, e0..7 = K 0..7, e8..15 = K 16..23
//                      lanes 16-31 -> M=lane-16, e0..7 = K 8..15, e8..15 = K 24..31
// ---------------------------------------------------------------------------
__device__ __forceinline__ v16bf load_fragA(const bf16_t* tile, int ldr, int lane) {
  const int grp = (lane >> 4) & 1;
  const int r   = lane & 15;
  const bf16_t* p = tile + (size_t)r * ldr + grp * 8;
  v8bf lo = *(const v8bf*)p;          // K: grp?8..15 : 0..7
  v8bf hi = *(const v8bf*)(p + 16);   // K: grp?24..31 : 16..23
  v16bf out;
#pragma unroll
  for (int i = 0; i < 8; ++i) { out[i] = lo[i]; out[i + 8] = hi[i]; }
  return out;
}

// B-matrix 32x16 bf16: lane holds column n = lane&15; lanes 0-15 -> e0..7 = K 0..7,
// e8..15 = K 16..23; lanes 16-31 -> e0..7 = K 8..15, e8..15 = K 24..31 (matches A).
__device__ __forceinline__ v16bf load_fragB(const bf16_t* colbase, int lane) {
  const int grp = (lane >> 4) & 1;
  const bf16_t* p = colbase + grp * 8;
  v8bf lo = *(const v8bf*)p;
  v8bf hi = *(const v8bf*)(p + 16);
  v16bf out;
#pragma unroll
  for (int i = 0; i < 8; ++i) { out[i] = lo[i]; out[i + 8] = hi[i]; }
  return out;
}

__device__ __forceinline__ v8f wmma_bf16(v16bf a, v16bf b, v8f c) {
  // D = A(16x32) * B(32x16) + C, f32 accumulate
  return __builtin_amdgcn_wmma_f32_16x16x32_bf16(false, a, false, b,
                                                 (short)0, c, false, false);
}

// ---------------------------------------------------------------------------
// K1: fp32 -> bf16 conversion
// ---------------------------------------------------------------------------
__global__ void cvt_f32_bf16_kernel(const float* __restrict__ src,
                                    bf16_t* __restrict__ dst, int n) {
  for (int i = blockIdx.x * blockDim.x + threadIdx.x; i < n;
       i += gridDim.x * blockDim.x)
    dst[i] = (bf16_t)src[i];
}

// ---------------------------------------------------------------------------
// K2: fused Q/K/V projection.  y[m,n] = sum_d x[m,d] * W[n,d] + bias[n]
// One wave computes a 32x32 output tile (2x2 WMMA accumulators).
// V is written transposed as Vt[b, n, s] for contiguous attn*V fragments.
// ---------------------------------------------------------------------------
__global__ void proj_qkv_kernel(const bf16_t* __restrict__ xb,
                                const bf16_t* __restrict__ wb,   // [3][D][D]
                                const float* __restrict__ bq,
                                const float* __restrict__ bk,
                                const float* __restrict__ bv,
                                bf16_t* __restrict__ Qb,
                                bf16_t* __restrict__ Kb,
                                bf16_t* __restrict__ Vtb) {
  const int lane  = threadIdx.x & 31;
  const int n0    = blockIdx.x * 32;
  const int m0    = blockIdx.y * 32;
  const int which = blockIdx.z;
  const bf16_t* W = wb + (size_t)which * DMODEL * DMODEL;
  const float* bias = (which == 0) ? bq : (which == 1) ? bk : bv;
  const int col = lane & 15;
  const int grp = (lane >> 4) & 1;

  v8f acc[2][2];
#pragma unroll
  for (int i = 0; i < 2; ++i)
#pragma unroll
    for (int j = 0; j < 2; ++j)
#pragma unroll
      for (int e = 0; e < 8; ++e) acc[i][j][e] = 0.f;

  for (int k0 = 0; k0 < DMODEL; k0 += 32) {
    __builtin_prefetch(xb + (size_t)m0 * DMODEL + k0 + 64, 0, 1);
    v16bf a0 = load_fragA(xb + (size_t)m0 * DMODEL + k0, DMODEL, lane);
    v16bf a1 = load_fragA(xb + (size_t)(m0 + 16) * DMODEL + k0, DMODEL, lane);
    v16bf b0 = load_fragB(W + (size_t)(n0 + col) * DMODEL + k0, lane);
    v16bf b1 = load_fragB(W + (size_t)(n0 + 16 + col) * DMODEL + k0, lane);
    acc[0][0] = wmma_bf16(a0, b0, acc[0][0]);
    acc[0][1] = wmma_bf16(a0, b1, acc[0][1]);
    acc[1][0] = wmma_bf16(a1, b0, acc[1][0]);
    acc[1][1] = wmma_bf16(a1, b1, acc[1][1]);
  }

#pragma unroll
  for (int mi = 0; mi < 2; ++mi) {
#pragma unroll
    for (int ni = 0; ni < 2; ++ni) {
      const int n = n0 + ni * 16 + col;
      const float bval = bias[n];
#pragma unroll
      for (int r = 0; r < 8; ++r) {
        const int m = m0 + mi * 16 + r + grp * 8;   // token index
        const bf16_t hv = (bf16_t)(acc[mi][ni][r] + bval);
        if (which == 0)      Qb[(size_t)m * DMODEL + n] = hv;
        else if (which == 1) Kb[(size_t)m * DMODEL + n] = hv;
        else {
          const int bb = m / SLEN, s = m % SLEN;
          Vtb[((size_t)bb * DMODEL + n) * SLEN + s] = hv;   // [B, HDk, S]
        }
      }
    }
  }
}

// ---------------------------------------------------------------------------
// K3: flash attention, one wave = 16 queries of one head, online softmax.
// C-layout rows live in 16-lane groups -> shfl_xor width 16 row reductions.
// P tile goes through LDS to change C-layout -> A-fragment layout.
// ---------------------------------------------------------------------------
__global__ void flash_attn_kernel(const bf16_t* __restrict__ Qb,
                                  const bf16_t* __restrict__ Kb,
                                  const bf16_t* __restrict__ Vtb,
                                  bf16_t* __restrict__ ctxb) {
  __shared__ __align__(16) bf16_t pshare[4][16 * 32];
  const int lane = threadIdx.x & 31;
  const int wave = threadIdx.x >> 5;
  const int bh = blockIdx.y;
  const int b  = bh >> 4;
  const int h  = bh & 15;
  const int q0 = blockIdx.x * 64 + wave * 16;
  const int col = lane & 15;
  const int grp = (lane >> 4) & 1;

  const bf16_t* Qbase = Qb  + (size_t)b * SLEN * DMODEL + h * DK;
  const bf16_t* Kbase = Kb  + (size_t)b * SLEN * DMODEL + h * DK;
  const bf16_t* Vbase = Vtb + ((size_t)b * DMODEL + h * DK) * SLEN;

  // Q fragments (dk halves), fixed for the whole key loop
  v16bf aQ0 = load_fragA(Qbase + (size_t)q0 * DMODEL + 0,  DMODEL, lane);
  v16bf aQ1 = load_fragA(Qbase + (size_t)q0 * DMODEL + 32, DMODEL, lane);

  v8f acc[4];
  float mrow[8], lrow[8];
#pragma unroll
  for (int t = 0; t < 4; ++t)
#pragma unroll
    for (int e = 0; e < 8; ++e) acc[t][e] = 0.f;
#pragma unroll
  for (int r = 0; r < 8; ++r) { mrow[r] = -3.0e38f; lrow[r] = 0.f; }

  bf16_t* pl = &pshare[wave][0];
  const float scale = 0.125f;   // 1/sqrt(64)

  for (int key0 = 0; key0 < SLEN; key0 += 32) {
    v8f c0, c1;
#pragma unroll
    for (int e = 0; e < 8; ++e) { c0[e] = 0.f; c1[e] = 0.f; }
    // scores: 16 queries x 32 keys, K-dim = dk = 64 (2 WMMA steps each)
    c0 = wmma_bf16(aQ0, load_fragB(Kbase + (size_t)(key0 + col) * DMODEL + 0,  lane), c0);
    c0 = wmma_bf16(aQ1, load_fragB(Kbase + (size_t)(key0 + col) * DMODEL + 32, lane), c0);
    c1 = wmma_bf16(aQ0, load_fragB(Kbase + (size_t)(key0 + 16 + col) * DMODEL + 0,  lane), c1);
    c1 = wmma_bf16(aQ1, load_fragB(Kbase + (size_t)(key0 + 16 + col) * DMODEL + 32, lane), c1);

#pragma unroll
    for (int r = 0; r < 8; ++r) {
      float s0 = c0[r] * scale;
      float s1 = c1[r] * scale;
      float mx = fmaxf(s0, s1);
#pragma unroll
      for (int off = 8; off >= 1; off >>= 1)
        mx = fmaxf(mx, __shfl_xor(mx, off, 16));
      const float mn = fmaxf(mrow[r], mx);
      const float al = __expf(mrow[r] - mn);
      const float p0 = __expf(s0 - mn);
      const float p1 = __expf(s1 - mn);
      float rs = p0 + p1;
#pragma unroll
      for (int off = 8; off >= 1; off >>= 1)
        rs += __shfl_xor(rs, off, 16);
      lrow[r] = lrow[r] * al + rs;
      mrow[r] = mn;
#pragma unroll
      for (int t = 0; t < 4; ++t) acc[t][r] *= al;
      const int row = r + grp * 8;
      pl[row * 32 + col]      = (bf16_t)p0;
      pl[row * 32 + col + 16] = (bf16_t)p1;
    }
    // fence LDS stores before re-reading P in A-fragment layout (same wave,
    // split DS counter on CDNA5)
    asm volatile("s_wait_dscnt 0" ::: "memory");
    v16bf aP = load_fragA(pl, 32, lane);
#pragma unroll
    for (int t = 0; t < 4; ++t) {
      const bf16_t* vcol = Vbase + (size_t)(t * 16 + col) * SLEN + key0;
      acc[t] = wmma_bf16(aP, load_fragB(vcol, lane), acc[t]);
    }
  }

#pragma unroll
  for (int t = 0; t < 4; ++t) {
#pragma unroll
    for (int r = 0; r < 8; ++r) {
      const int row = r + grp * 8;
      const float v = acc[t][r] / lrow[r];
      const size_t tok = (size_t)b * SLEN + q0 + row;
      ctxb[tok * DMODEL + h * DK + t * 16 + col] = (bf16_t)v;
    }
  }
}

// ---------------------------------------------------------------------------
// K4a: output projection + bias + residual, fp32 result
// ---------------------------------------------------------------------------
__global__ void outproj_kernel(const bf16_t* __restrict__ ctxb,
                               const bf16_t* __restrict__ Wob,
                               const float* __restrict__ bo,
                               const float* __restrict__ x,
                               float* __restrict__ y) {
  const int lane = threadIdx.x & 31;
  const int n0 = blockIdx.x * 32;
  const int m0 = blockIdx.y * 32;
  const int col = lane & 15;
  const int grp = (lane >> 4) & 1;

  v8f acc[2][2];
#pragma unroll
  for (int i = 0; i < 2; ++i)
#pragma unroll
    for (int j = 0; j < 2; ++j)
#pragma unroll
      for (int e = 0; e < 8; ++e) acc[i][j][e] = 0.f;

  for (int k0 = 0; k0 < DMODEL; k0 += 32) {
    v16bf a0 = load_fragA(ctxb + (size_t)m0 * DMODEL + k0, DMODEL, lane);
    v16bf a1 = load_fragA(ctxb + (size_t)(m0 + 16) * DMODEL + k0, DMODEL, lane);
    v16bf b0 = load_fragB(Wob + (size_t)(n0 + col) * DMODEL + k0, lane);
    v16bf b1 = load_fragB(Wob + (size_t)(n0 + 16 + col) * DMODEL + k0, lane);
    acc[0][0] = wmma_bf16(a0, b0, acc[0][0]);
    acc[0][1] = wmma_bf16(a0, b1, acc[0][1]);
    acc[1][0] = wmma_bf16(a1, b0, acc[1][0]);
    acc[1][1] = wmma_bf16(a1, b1, acc[1][1]);
  }

#pragma unroll
  for (int mi = 0; mi < 2; ++mi) {
#pragma unroll
    for (int ni = 0; ni < 2; ++ni) {
      const int n = n0 + ni * 16 + col;
      const float bval = bo[n];
#pragma unroll
      for (int r = 0; r < 8; ++r) {
        const int m = m0 + mi * 16 + r + grp * 8;
        y[(size_t)m * DMODEL + n] =
            acc[mi][ni][r] + bval + x[(size_t)m * DMODEL + n];
      }
    }
  }
}

// ---------------------------------------------------------------------------
// K4b: LayerNorm, one block (256 threads = 8 waves) per token row
// ---------------------------------------------------------------------------
__global__ void layernorm_kernel(const float* __restrict__ y,
                                 const float* __restrict__ gamma,
                                 const float* __restrict__ beta,
                                 float* __restrict__ out) {
  __shared__ float ssum[8], ssq[8];
  const int row = blockIdx.x;
  const float* yr = y + (size_t)row * DMODEL;
  float s = 0.f, sq = 0.f;
  for (int i = threadIdx.x; i < DMODEL; i += 256) {
    const float v = yr[i];
    s += v; sq += v * v;
  }
#pragma unroll
  for (int off = 16; off >= 1; off >>= 1) {
    s  += __shfl_xor(s,  off, 32);
    sq += __shfl_xor(sq, off, 32);
  }
  const int wave = threadIdx.x >> 5, lane = threadIdx.x & 31;
  if (lane == 0) { ssum[wave] = s; ssq[wave] = sq; }
  __syncthreads();
  float ts = 0.f, tq = 0.f;
#pragma unroll
  for (int w = 0; w < 8; ++w) { ts += ssum[w]; tq += ssq[w]; }
  const float mean = ts * (1.0f / DMODEL);
  const float var  = tq * (1.0f / DMODEL) - mean * mean;
  const float inv  = rsqrtf(var + 1e-5f);
  float* outr = out + (size_t)row * DMODEL;
  for (int i = threadIdx.x; i < DMODEL; i += 256)
    outr[i] = (yr[i] - mean) * inv * gamma[i] + beta[i];
}

// ---------------------------------------------------------------------------
// Host-side launch
// ---------------------------------------------------------------------------
extern "C" void kernel_launch(void* const* d_in, const int* in_sizes, int n_in,
                              void* d_out, int out_size, void* d_ws, size_t ws_size,
                              hipStream_t stream) {
  const float* x  = (const float*)d_in[0];
  const float* Wq = (const float*)d_in[1];
  const float* bq = (const float*)d_in[2];
  const float* Wk = (const float*)d_in[3];
  const float* bk = (const float*)d_in[4];
  const float* Wv = (const float*)d_in[5];
  const float* bv = (const float*)d_in[6];
  const float* Wo = (const float*)d_in[7];
  const float* bo = (const float*)d_in[8];
  const float* g  = (const float*)d_in[9];
  const float* be = (const float*)d_in[10];
  float* out = (float*)d_out;

  const size_t MB = (size_t)1024 * 1024;
  char* ws = (char*)d_ws;
  bf16_t* xb   = (bf16_t*)(ws);               // 16 MB  [K1 -> K2]
  bf16_t* wb   = (bf16_t*)(ws + 16 * MB);     //  8 MB  [K1 -> K2, K4a]
  bf16_t* Qb   = (bf16_t*)(ws + 24 * MB);     // 16 MB  [K2 -> K3]
  bf16_t* Kb   = (bf16_t*)(ws + 40 * MB);     // 16 MB  [K2 -> K3]
  bf16_t* Vtb  = (bf16_t*)(ws + 56 * MB);     // 16 MB  [K2 -> K3]
  bf16_t* ctxb = (bf16_t*)(ws + 72 * MB);     // 16 MB  [K3 -> K4a]
  float*  y    = (float*)(ws + 24 * MB);      // 32 MB  [K4a -> K4b], overlays dead Qb/Kb

  const int nx = MTOT * DMODEL;
  const int nw = DMODEL * DMODEL;
  cvt_f32_bf16_kernel<<<2048, 256, 0, stream>>>(x,  xb,              nx);
  cvt_f32_bf16_kernel<<<1024, 256, 0, stream>>>(Wq, wb + 0 * (size_t)nw, nw);
  cvt_f32_bf16_kernel<<<1024, 256, 0, stream>>>(Wk, wb + 1 * (size_t)nw, nw);
  cvt_f32_bf16_kernel<<<1024, 256, 0, stream>>>(Wv, wb + 2 * (size_t)nw, nw);
  cvt_f32_bf16_kernel<<<1024, 256, 0, stream>>>(Wo, wb + 3 * (size_t)nw, nw);

  proj_qkv_kernel<<<dim3(DMODEL / 32, MTOT / 32, 3), 32, 0, stream>>>(
      xb, wb, bq, bk, bv, Qb, Kb, Vtb);

  flash_attn_kernel<<<dim3(SLEN / 64, BATCH * NHEADS), 128, 0, stream>>>(
      Qb, Kb, Vtb, ctxb);

  outproj_kernel<<<dim3(DMODEL / 32, MTOT / 32), 32, 0, stream>>>(
      ctxb, wb + 3 * (size_t)nw, bo, x, y);

  layernorm_kernel<<<MTOT, 256, 0, stream>>>(y, g, be, out);
}
